// V18AlgebraMultistepModelA_ActionZ_61340722921655
// MI455X (gfx1250) — matrix operand order repaired and checked by reading, not compiled, not run on hardware
//
#include <hip/hip_runtime.h>

// ---------------------------------------------------------------------------
// Problem constants (match reference): B, V, N, Y, HID, T = 512,16,8192,32,512,8
// ---------------------------------------------------------------------------
constexpr int Bc = 512, Vc = 16, Nc = 8192, Yc = 32, HIDc = 512, Tc = 8;
constexpr int KPAD = 64;   // feat (52) and fy (48) padded to 64 for K%32==0

typedef _Float16 h16;
typedef __attribute__((ext_vector_type(16))) _Float16 v16h;
typedef __attribute__((ext_vector_type(8)))  _Float16 v8h;
typedef __attribute__((ext_vector_type(8)))  float    v8f;

// ---------------------------------------------------------------------------
// Kernel 1: per-batch scan. One workgroup per batch row.
//   - collapse base/action constraints into per-view need[] (+contradiction)
//   - single pass over all 16 views with b128 loads
//   - LDS: sigma histogram counts + per-view 32-bit label masks
//   - thread 0 finalizes: p_sig, uniq=popcount, entropy, top2, mass -> feat f16
// ---------------------------------------------------------------------------
__global__ __launch_bounds__(256) void scan_feat_kernel(
    const int* __restrict__ tables, const int* __restrict__ sigma,
    const int* __restrict__ base_obs, const int* __restrict__ actions,
    const int* __restrict__ responses, const int* __restrict__ tval,
    h16* __restrict__ feat_h)            // [B, 64] f16, rows 52..63 zero
{
    const int b   = blockIdx.x;
    const int tid = threadIdx.x;

    __shared__ int      s_need[Vc];
    __shared__ int      s_contra;
    __shared__ unsigned s_mask[Vc];
    __shared__ int      s_cnt[Yc];
    __shared__ float    s_feat[KPAD];

    if (tid < Vc) { s_need[tid] = -1; s_mask[tid] = 0u; }
    if (tid < Yc) s_cnt[tid] = 0;
    if (tid == 0) s_contra = 0;
    __syncthreads();

    if (tid == 0) {
        s_need[0] = base_obs[b];
        const int tt = tval[0];
        for (int j = 0; j < tt; ++j) {
            const int v = actions[b * Tc + j];
            const int r = responses[b * Tc + j];
            if (s_need[v] < 0)          s_need[v] = r;
            else if (s_need[v] != r)    s_contra = 1;
        }
    }
    __syncthreads();

    const int contra = s_contra;
    int need[Vc];
#pragma unroll
    for (int v = 0; v < Vc; ++v) need[v] = s_need[v];

    unsigned lmask[Vc];
#pragma unroll
    for (int v = 0; v < Vc; ++v) lmask[v] = 0u;

    const int4* tabv = reinterpret_cast<const int4*>(tables + (size_t)b * Vc * Nc);
    const int4* sigv = reinterpret_cast<const int4*>(sigma  + (size_t)b * Nc);
    constexpr int NV = Nc / 4;   // 2048 int4 per view

    for (int q = tid; q < NV; q += 256) {
        __builtin_prefetch(&tabv[q + 256], 0, 1);   // global_prefetch, next tile
        int4 tv[Vc];
#pragma unroll
        for (int v = 0; v < Vc; ++v) tv[v] = tabv[v * NV + q];

        unsigned ok = contra ? 0u : 0xFu;
#pragma unroll
        for (int v = 0; v < Vc; ++v) {
            const int nv = need[v];
            if (nv >= 0) {
                unsigned mm = 0u;
                mm |= (tv[v].x == nv) ? 1u : 0u;
                mm |= (tv[v].y == nv) ? 2u : 0u;
                mm |= (tv[v].z == nv) ? 4u : 0u;
                mm |= (tv[v].w == nv) ? 8u : 0u;
                ok &= mm;
            }
        }
        if (ok) {
            const int4 sg = sigv[q];
            const int svals[4] = { sg.x, sg.y, sg.z, sg.w };
#pragma unroll
            for (int e = 0; e < 4; ++e) {
                if (ok & (1u << e)) {
                    atomicAdd(&s_cnt[svals[e] & (Yc - 1)], 1);
#pragma unroll
                    for (int v = 0; v < Vc; ++v) {
                        const int lab = (e == 0) ? tv[v].x : (e == 1) ? tv[v].y
                                       : (e == 2) ? tv[v].z : tv[v].w;
                        lmask[v] |= 1u << (lab & 31);
                    }
                }
            }
        }
    }
#pragma unroll
    for (int v = 0; v < Vc; ++v)
        if (lmask[v]) atomicOr(&s_mask[v], lmask[v]);
    __syncthreads();

    if (tid == 0) {
        int total = 0;
        for (int y = 0; y < Yc; ++y) total += s_cnt[y];
        const float inv = 1.0f / (float)(total > 0 ? total : 1);
        float ent = 0.0f, m1 = -1.0f, m2 = -1.0f;
        for (int y = 0; y < Yc; ++y) {
            const float p = (float)s_cnt[y] * inv;
            s_feat[y] = p;
            const float pc = fmaxf(p, 1e-9f);
            ent -= pc * logf(pc);
            if (p > m1)      { m2 = m1; m1 = p; }
            else if (p > m2) { m2 = p; }
        }
        for (int v = 0; v < Vc; ++v) s_feat[Yc + v] = (float)__popc(s_mask[v]);
        s_feat[48] = ent;
        s_feat[49] = m1;
        s_feat[50] = m2;
        s_feat[51] = (total > 0) ? inv : 0.0f;     // max(state_dist)
        for (int i = 52; i < KPAD; ++i) s_feat[i] = 0.0f;
    }
    __syncthreads();
    if (tid < KPAD) feat_h[(size_t)b * KPAD + tid] = (h16)s_feat[tid];
}

// ---------------------------------------------------------------------------
// Kernel 2: fp32 -> f16 weight staging, stored TRANSPOSED [N, K] so the
// WMMA B-fragment (lane = column, two runs of 8 K, 16 apart) is two b128 loads.
// ---------------------------------------------------------------------------
__global__ __launch_bounds__(256) void convert_weights_kernel(
    const float* __restrict__ W1, const float* __restrict__ Wy1,
    const float* __restrict__ Wy2,
    h16* __restrict__ W1t,   // [HID, KPAD]
    h16* __restrict__ Wy1t,  // [HID, KPAD]
    h16* __restrict__ Wy2t)  // [Y,   HID]
{
    const int idx = blockIdx.x * blockDim.x + threadIdx.x;
    if (idx < KPAD * HIDc) {
        const int row = idx / HIDc, col = idx % HIDc;   // row = K, col = N
        W1t [(size_t)col * KPAD + row] =
            (h16)((row < Yc + Vc + 4) ? W1[row * HIDc + col]  : 0.0f);
        Wy1t[(size_t)col * KPAD + row] =
            (h16)((row < Yc + Vc)     ? Wy1[row * HIDc + col] : 0.0f);
    }
    if (idx < HIDc * Yc) {
        const int row = idx / Yc, col = idx % Yc;       // row = K, col = N
        Wy2t[(size_t)col * HIDc + row] = (h16)Wy2[idx];
    }
}

// ---------------------------------------------------------------------------
// WMMA fragment loaders, per CDNA5 ISA 7.12.2 (wave32, 16-bit operands)
// A 16x32 (MxK): lanes 0-15 row=lane, halves = K{0..7,16..23};
//                lanes 16-31 row=lane-16, halves = K{8..15,24..31}
// B 32x16 (KxN): lane%16 = column; lanes 0-15 hold K{0..7,16..23}, 16-31 hold K{8..15,24..31}
// C/D 16x16 f32: lane%16 = N; vgpr r = M=r (lanes 0-15) / M=r+8 (lanes 16-31)
// Both loaders emit two global_load_b128 per fragment.
// ---------------------------------------------------------------------------
__device__ inline v16h load_a16(const h16* __restrict__ A, int lda, int row0, int k0)
{
    const int lane = threadIdx.x & 31;
    const h16* p = A + (size_t)(row0 + (lane & 15)) * lda + k0 + ((lane >> 4) << 3);
    const v8h lo = *reinterpret_cast<const v8h*>(p);        // K+0..7  (or 8..15)
    const v8h hi = *reinterpret_cast<const v8h*>(p + 16);   // K+16..23 (or 24..31)
    v16h a;
#pragma unroll
    for (int i = 0; i < 8; ++i) { a[i] = lo[i]; a[8 + i] = hi[i]; }
    return a;
}

__device__ inline v16h load_bt16(const h16* __restrict__ Bt, int ldk, int k0, int col0)
{
    const int lane = threadIdx.x & 31;
    const h16* p = Bt + (size_t)(col0 + (lane & 15)) * ldk + k0 + ((lane >> 4) << 3);
    const v8h lo = *reinterpret_cast<const v8h*>(p);        // K+0..7  (or 8..15)
    const v8h hi = *reinterpret_cast<const v8h*>(p + 16);   // K+16..23 (or 24..31)
    v16h b;
#pragma unroll
    for (int i = 0; i < 8; ++i) { b[i] = lo[i]; b[8 + i] = hi[i]; }
    return b;
}

// ---------------------------------------------------------------------------
// C = act(A[MxK]f16 * Bt[NxK]f16 + bias); K compile-time so WMMA chain unrolls.
// One wave per 16x16 output tile.
// ---------------------------------------------------------------------------
template <int K, bool GELU>
__global__ __launch_bounds__(32) void wmma_gemm_kernel(
    const h16* __restrict__ A, const h16* __restrict__ Bt,
    const float* __restrict__ bias,
    h16* __restrict__ OutH, float* __restrict__ OutF,
    int lda, int ldo)
{
    const int row0 = blockIdx.x * 16;
    const int col0 = blockIdx.y * 16;
    v8f c = {0.f, 0.f, 0.f, 0.f, 0.f, 0.f, 0.f, 0.f};
#pragma unroll
    for (int k0 = 0; k0 < K; k0 += 32) {
        const v16h a = load_a16(A, lda, row0, k0);
        const v16h b = load_bt16(Bt, K, k0, col0);
        c = __builtin_amdgcn_wmma_f32_16x16x32_f16(false, a, false, b,
                                                   (short)0, c, false, false);
    }
    const int lane = threadIdx.x & 31;
    const int col  = col0 + (lane & 15);
    const int rb   = row0 + ((lane >> 4) << 3);
    const float bz = bias[col];
#pragma unroll
    for (int r = 0; r < 8; ++r) {
        const float x = c[r] + bz;
        if (GELU) {
            const float g = 0.5f * x * (1.0f + erff(x * 0.70710678118654752f));
            OutH[(size_t)(rb + r) * ldo + col] = (h16)g;
        } else {
            OutF[(size_t)(rb + r) * ldo + col] = x;
        }
    }
}

// ---------------------------------------------------------------------------
// Kernel 4: z_logits = h @ W2 + b2 (tiny, scalar per row), softmax/argmax,
// straight-through one-hot, build fy = [p_sig(32) | z(16) | 0 pad]
// ---------------------------------------------------------------------------
__global__ __launch_bounds__(256) void z_fy_kernel(
    const h16* __restrict__ Hh, const float* __restrict__ W2,
    const float* __restrict__ b2, const h16* __restrict__ feat_h,
    h16* __restrict__ fy_h)
{
    const int b = blockIdx.x * blockDim.x + threadIdx.x;
    if (b >= Bc) return;
    float acc[Vc];
#pragma unroll
    for (int v = 0; v < Vc; ++v) acc[v] = b2[v];
    const h16* hrow = Hh + (size_t)b * HIDc;
    for (int j = 0; j < HIDc; ++j) {
        const float hj = (float)hrow[j];
#pragma unroll
        for (int v = 0; v < Vc; ++v) acc[v] += hj * W2[j * Vc + v];
    }
    float mx = acc[0]; int arg = 0;
#pragma unroll
    for (int v = 1; v < Vc; ++v) if (acc[v] > mx) { mx = acc[v]; arg = v; }
    float e[Vc], sum = 0.0f;
#pragma unroll
    for (int v = 0; v < Vc; ++v) { e[v] = expf(acc[v] - mx); sum += e[v]; }
    const float invs = 1.0f / sum;

    h16* out = fy_h + (size_t)b * KPAD;
    const h16* fr = feat_h + (size_t)b * KPAD;
    for (int y = 0; y < Yc; ++y) out[y] = fr[y];          // p_sig
#pragma unroll
    for (int v = 0; v < Vc; ++v) {
        const float s  = e[v] * invs;
        const float zh = (v == arg) ? 1.0f : 0.0f;
        const float z  = (zh - s) + s;                    // straight-through fwd
        out[Yc + v] = (h16)z;
    }
    for (int i = Yc + Vc; i < KPAD; ++i) out[i] = (h16)0.0f;
}

// ---------------------------------------------------------------------------
// Host launcher
// ---------------------------------------------------------------------------
extern "C" void kernel_launch(void* const* d_in, const int* in_sizes, int n_in,
                              void* d_out, int out_size, void* d_ws, size_t ws_size,
                              hipStream_t stream)
{
    const int*   tables    = (const int*)d_in[0];
    const int*   sigma     = (const int*)d_in[1];
    const int*   base_obs  = (const int*)d_in[2];
    const int*   actions   = (const int*)d_in[3];
    const int*   responses = (const int*)d_in[4];
    const int*   tval      = (const int*)d_in[5];
    const float* W1        = (const float*)d_in[6];
    const float* b1        = (const float*)d_in[7];
    const float* W2        = (const float*)d_in[8];
    const float* b2        = (const float*)d_in[9];
    const float* Wy1       = (const float*)d_in[10];
    const float* by1       = (const float*)d_in[11];
    const float* Wy2       = (const float*)d_in[12];
    const float* by2       = (const float*)d_in[13];
    float* out = (float*)d_out;

    // workspace partition (all f16 buffers), 256B aligned slices
    char* ws = (char*)d_ws;
    size_t off = 0;
    auto alloc = [&](size_t bytes) {
        char* p = ws + off;
        off += (bytes + 255) & ~(size_t)255;
        return p;
    };
    h16* feat_h = (h16*)alloc((size_t)Bc * KPAD * sizeof(h16));
    h16* W1t    = (h16*)alloc((size_t)HIDc * KPAD * sizeof(h16));
    h16* Wy1t   = (h16*)alloc((size_t)HIDc * KPAD * sizeof(h16));
    h16* Wy2t   = (h16*)alloc((size_t)Yc * HIDc * sizeof(h16));
    h16* h_h    = (h16*)alloc((size_t)Bc * HIDc * sizeof(h16));
    h16* fy_h   = (h16*)alloc((size_t)Bc * KPAD * sizeof(h16));
    h16* hy_h   = (h16*)alloc((size_t)Bc * HIDc * sizeof(h16));
    (void)ws_size; (void)n_in; (void)in_sizes; (void)out_size;

    // 1) bandwidth-bound scan -> feat (f16, K-padded)
    scan_feat_kernel<<<Bc, 256, 0, stream>>>(tables, sigma, base_obs, actions,
                                             responses, tval, feat_h);
    // 2) stage weights as f16, transposed for contiguous B-fragment loads
    convert_weights_kernel<<<(KPAD * HIDc + 255) / 256, 256, 0, stream>>>(
        W1, Wy1, Wy2, W1t, Wy1t, Wy2t);
    // 3) h = gelu(feat @ W1 + b1)          [512,64]x[64,512] -> f16
    wmma_gemm_kernel<KPAD, true><<<dim3(Bc / 16, HIDc / 16), 32, 0, stream>>>(
        feat_h, W1t, b1, h_h, nullptr, KPAD, HIDc);
    // 4) z_logits = h @ W2 + b2 ; softmax/argmax ; fy = [p_sig | z]
    z_fy_kernel<<<Bc / 256, 256, 0, stream>>>(h_h, W2, b2, feat_h, fy_h);
    // 5) hy = gelu(fy @ Wy1 + by1)         [512,64]x[64,512] -> f16
    wmma_gemm_kernel<KPAD, true><<<dim3(Bc / 16, HIDc / 16), 32, 0, stream>>>(
        fy_h, Wy1t, by1, hy_h, nullptr, KPAD, HIDc);
    // 6) out = hy @ Wy2 + by2              [512,512]x[512,32] -> fp32
    wmma_gemm_kernel<HIDc, false><<<dim3(Bc / 16, Yc / 16), 32, 0, stream>>>(
        hy_h, Wy2t, by2, nullptr, out, HIDc, Yc);
}
